// Triangle_Att_Start_41583873360494
// MI455X (gfx1250) — matrix operand order repaired and checked by reading, not compiled, not run on hardware
//
#include <hip/hip_runtime.h>
#include <hip/hip_bf16.h>

// ---------------------------------------------------------------------------
// Triangle attention (starting node), N=256, C=128, H=4, D=32, all GEMMs on
// v_wmma_f32_16x16x32_bf16.
// ---------------------------------------------------------------------------

typedef __bf16 bf16;
typedef __attribute__((ext_vector_type(4)))  bf16  v4bf;
typedef __attribute__((ext_vector_type(8)))  bf16  v8bf;
typedef __attribute__((ext_vector_type(16))) bf16  v16bf;
typedef __attribute__((ext_vector_type(8)))  float v8f;

#define NPOS 256
#define CDIM 128
#define HH 4
#define DD 32
#define NNPOS 65536   // NPOS*NPOS
#define QSCALE 0.17677669529663687f   // 1/sqrt(32)

__device__ __forceinline__ bf16 f2bf(float f) {
  unsigned u = __builtin_bit_cast(unsigned, f);
  unsigned r = (u + 0x7FFFu + ((u >> 16) & 1u)) >> 16;
  unsigned short s = (unsigned short)r;
  return __builtin_bit_cast(bf16, s);
}
__device__ __forceinline__ float bf2f(bf16 h) {
  unsigned short s = __builtin_bit_cast(unsigned short, h);
  unsigned u = ((unsigned)s) << 16;
  return __builtin_bit_cast(float, u);
}

__device__ __forceinline__ v16bf cat8(v8bf lo, v8bf hi) {
  return __builtin_shufflevector(lo, hi, 0,1,2,3,4,5,6,7,8,9,10,11,12,13,14,15);
}

// A fragment (16x32, bf16): lane holds row M = lane%16; k chunks at
// {g*8 .. g*8+7} and {16+g*8 .. 16+g*8+7}, g = lane/16.  base = &A[row0][k0].
__device__ __forceinline__ v16bf load_a16(const bf16* base, int row_stride, int lane) {
  int r = lane & 15, g = lane >> 4;
  const bf16* p = base + (size_t)r * row_stride + g * 8;
  v8bf lo = *(const v8bf*)(p);
  v8bf hi = *(const v8bf*)(p + 16);
  return cat8(lo, hi);
}

// B fragment (32x16, bf16): lane holds column N = lane%16; k chunks at
// {g*8 .. g*8+7} and {16+g*8 .. 16+g*8+7}.  Memory must be laid out so "N" selects a row with K contiguous
// (i.e. B stored N-major), base = &B[n0][k0].
__device__ __forceinline__ v16bf load_b16(const bf16* base, int row_stride, int lane) {
  int n = lane & 15, g = lane >> 4;
  const bf16* p = base + (size_t)n * row_stride + g * 8;
  v8bf lo = *(const v8bf*)(p);
  v8bf hi = *(const v8bf*)(p + 16);
  return cat8(lo, hi);
}

__device__ __forceinline__ v8f wmma_bf(v16bf a, v16bf b, v8f c) {
  return __builtin_amdgcn_wmma_f32_16x16x32_bf16(
      /*neg_a=*/false, a, /*neg_b=*/false, b,
      /*c_mod=*/(short)0, c, /*reuse_a=*/false, /*reuse_b=*/false);
}

// ---------------------------------------------------------------------------
// Kernel 1: LayerNorm (+ pair bias z@Wb).  One wave per position (wave32,
// 4 channels/lane).  Writes z as bf16 [m][128], bias as fp32 [m][4].
// ---------------------------------------------------------------------------
__global__ __launch_bounds__(256) void ln_kernel(
    const float* __restrict__ pair, const float* __restrict__ lns,
    const float* __restrict__ lnb, const float* __restrict__ Wb,
    bf16* __restrict__ zb, float* __restrict__ biasb) {
  int wid = threadIdx.x >> 5, lane = threadIdx.x & 31;
  size_t m = (size_t)blockIdx.x * 8 + wid;
  const float* x = pair + m * CDIM;
  float4 v = *(const float4*)(x + lane * 4);

  float s = v.x + v.y + v.z + v.w;
  #pragma unroll
  for (int mk = 16; mk >= 1; mk >>= 1) s += __shfl_xor(s, mk);
  float mu = s * (1.0f / CDIM);

  float d0 = v.x - mu, d1 = v.y - mu, d2 = v.z - mu, d3 = v.w - mu;
  float vs = d0*d0 + d1*d1 + d2*d2 + d3*d3;
  #pragma unroll
  for (int mk = 16; mk >= 1; mk >>= 1) vs += __shfl_xor(vs, mk);
  float inv = rsqrtf(vs * (1.0f / CDIM) + 1e-5f);

  float4 sc = *(const float4*)(lns + lane * 4);
  float4 bi = *(const float4*)(lnb + lane * 4);
  float z0 = d0 * inv * sc.x + bi.x;
  float z1 = d1 * inv * sc.y + bi.y;
  float z2 = d2 * inv * sc.z + bi.z;
  float z3 = d3 * inv * sc.w + bi.w;

  v4bf zv;
  zv[0] = f2bf(z0); zv[1] = f2bf(z1); zv[2] = f2bf(z2); zv[3] = f2bf(z3);
  *(v4bf*)(zb + m * CDIM + lane * 4) = zv;

  // pair bias: 4 dot products of length 128
  float zl[4] = {z0, z1, z2, z3};
  float p0 = 0.f, p1 = 0.f, p2 = 0.f, p3 = 0.f;
  #pragma unroll
  for (int c = 0; c < 4; ++c) {
    const float* wr = Wb + (size_t)(lane * 4 + c) * HH;
    p0 += zl[c] * wr[0]; p1 += zl[c] * wr[1];
    p2 += zl[c] * wr[2]; p3 += zl[c] * wr[3];
  }
  #pragma unroll
  for (int mk = 16; mk >= 1; mk >>= 1) {
    p0 += __shfl_xor(p0, mk); p1 += __shfl_xor(p1, mk);
    p2 += __shfl_xor(p2, mk); p3 += __shfl_xor(p3, mk);
  }
  if (lane == 0) {
    float4 bv; bv.x = p0; bv.y = p1; bv.z = p2; bv.w = p3;
    *(float4*)(biasb + m * HH) = bv;
  }
}

// ---------------------------------------------------------------------------
// Kernel 2: transpose + bf16-convert the 5 128x128 weight matrices into
// Wt[n][k] so WMMA B-fragments load contiguously.  Order: q,k,v,g,o.
// ---------------------------------------------------------------------------
__global__ __launch_bounds__(128) void wt_kernel(
    const float* __restrict__ Wq, const float* __restrict__ Wk,
    const float* __restrict__ Wv, const float* __restrict__ Wg,
    const float* __restrict__ Wo, bf16* __restrict__ wT) {
  int b = blockIdx.x;
  int mat = b >> 7, n = b & 127, k = threadIdx.x;
  const float* W = (mat == 0) ? Wq : (mat == 1) ? Wk : (mat == 2) ? Wv
                  : (mat == 3) ? Wg : Wo;
  wT[(size_t)mat * CDIM * CDIM + (size_t)n * CDIM + k] = f2bf(W[(size_t)k * CDIM + n]);
}

// ---------------------------------------------------------------------------
// Kernel 3: projections  C[65536,128] = z @ W  with WMMA.
// blockIdx.y selects: 0=q (scaled), 1=k, 2=v (stored transposed), 3=gate.
// Block = 256 thr (8 waves): 64 M-rows x 128 N-cols; wave = 16x64.
// ---------------------------------------------------------------------------
__global__ __launch_bounds__(256) void proj_kernel(
    const bf16* __restrict__ zb, const bf16* __restrict__ wT,
    const float* __restrict__ bg, bf16* __restrict__ qb, bf16* __restrict__ kb,
    bf16* __restrict__ vT, bf16* __restrict__ gb) {
  int mode = blockIdx.y;
  const bf16* W = wT + (size_t)mode * CDIM * CDIM;
  int tid = threadIdx.x, wid = tid >> 5, lane = tid & 31;
  int mw = wid & 3, nw = wid >> 2;
  size_t mt = (size_t)blockIdx.x * 64 + (size_t)mw * 16;
  int n0 = lane & 15, g = lane >> 4;

  v8f acc[4] = {};
  #pragma unroll
  for (int c = 0; c < 4; ++c) {
    v16bf a = load_a16(zb + mt * CDIM + c * 32, CDIM, lane);
    #pragma unroll
    for (int t = 0; t < 4; ++t) {
      int nb = nw * 64 + t * 16;
      v16bf b = load_b16(W + (size_t)nb * CDIM + c * 32, CDIM, lane);
      acc[t] = wmma_bf(a, b, acc[t]);
    }
  }

  if (mode == 0) {                       // q: scale by 1/sqrt(D)
    #pragma unroll
    for (int t = 0; t < 4; ++t) {
      int n = nw * 64 + t * 16 + n0;
      #pragma unroll
      for (int r = 0; r < 8; ++r)
        qb[(mt + r + 8 * g) * CDIM + n] = f2bf(acc[t][r] * QSCALE);
    }
  } else if (mode == 1) {                // k: plain
    #pragma unroll
    for (int t = 0; t < 4; ++t) {
      int n = nw * 64 + t * 16 + n0;
      #pragma unroll
      for (int r = 0; r < 8; ++r)
        kb[(mt + r + 8 * g) * CDIM + n] = f2bf(acc[t][r]);
    }
  } else if (mode == 2) {                // v: store transposed [i][h*32+d][key]
    int i = (int)(mt >> 8);
    int key0 = (int)(mt & 255) + 8 * g;  // C-layout rows are consecutive keys
    #pragma unroll
    for (int t = 0; t < 4; ++t) {
      int n = nw * 64 + t * 16 + n0;     // n = h*32+d
      v8bf pv;
      #pragma unroll
      for (int r = 0; r < 8; ++r) pv[r] = f2bf(acc[t][r]);
      *(v8bf*)(vT + ((size_t)(i * CDIM + n)) * NPOS + key0) = pv;
    }
  } else {                               // gate: sigmoid(x + bg)
    #pragma unroll
    for (int t = 0; t < 4; ++t) {
      int n = nw * 64 + t * 16 + n0;
      float bgn = bg[n];
      #pragma unroll
      for (int r = 0; r < 8; ++r) {
        float x = acc[t][r] + bgn;
        gb[(mt + r + 8 * g) * CDIM + n] = f2bf(1.0f / (1.0f + __expf(-x)));
      }
    }
  }
}

// ---------------------------------------------------------------------------
// Kernel 4: fused attention per (i, h, 32-query tile).  4 waves / block.
// GEMM1 (QK^T, one WMMA per 16x16 tile since K=D=32) -> LDS fp32, softmax
// (unnormalized P in bf16 LDS, 1/sum folded into output), GEMM2 (P@V),
// gate multiply, store gated output bf16 [m][128].
// ---------------------------------------------------------------------------
__global__ __launch_bounds__(128) void attn_kernel(
    const bf16* __restrict__ qb, const bf16* __restrict__ kb,
    const bf16* __restrict__ vT, const bf16* __restrict__ gb,
    const float* __restrict__ biasb, bf16* __restrict__ go) {
  __shared__ float S[32][NPOS];
  __shared__ bf16  P[32][NPOS];
  __shared__ float rinv[32];

  int b = blockIdx.x;
  int i = b >> 5, rem = b & 31, h = rem >> 3, jt = rem & 7;
  int tid = threadIdx.x, w = tid >> 5, lane = tid & 31;
  int n0 = lane & 15, g = lane >> 4;

  // ---- GEMM1: S = q @ k^T + bias ----
  const bf16* qbase = qb + ((size_t)(i * NPOS + jt * 32)) * CDIM + h * DD;
  v16bf a0 = load_a16(qbase, CDIM, lane);
  v16bf a1 = load_a16(qbase + 16 * CDIM, CDIM, lane);
  #pragma unroll
  for (int t = 0; t < 4; ++t) {
    int kbase = (w * 4 + t) * 16;
    v16bf bk = load_b16(kb + ((size_t)(i * NPOS + kbase)) * CDIM + h * DD, CDIM, lane);
    v8f c0 = {}, c1 = {};
    c0 = wmma_bf(a0, bk, c0);
    c1 = wmma_bf(a1, bk, c1);
    int kg = kbase + n0;
    #pragma unroll
    for (int r = 0; r < 8; ++r) {
      int j0 = r + 8 * g;              // 0..15 within tile
      int jg0 = jt * 32 + j0;          // global query row (bias is i-independent)
      int jg1 = jg0 + 16;
      float bi0 = biasb[((size_t)jg0 * NPOS + kg) * HH + h];
      float bi1 = biasb[((size_t)jg1 * NPOS + kg) * HH + h];
      S[j0][kg]      = c0[r] + bi0;
      S[j0 + 16][kg] = c1[r] + bi1;
    }
  }
  __syncthreads();

  // ---- softmax over keys (4 threads per row) ----
  {
    int row = tid >> 2, q4 = tid & 3, cb = q4 * 64;
    float mx = -3.0e38f;
    #pragma unroll 8
    for (int c = 0; c < 64; ++c) mx = fmaxf(mx, S[row][cb + c]);
    mx = fmaxf(mx, __shfl_xor(mx, 1));
    mx = fmaxf(mx, __shfl_xor(mx, 2));
    float sum = 0.f;
    #pragma unroll 8
    for (int c = 0; c < 64; ++c) {
      float e = __expf(S[row][cb + c] - mx);
      sum += e;
      P[row][cb + c] = f2bf(e);
    }
    sum += __shfl_xor(sum, 1);
    sum += __shfl_xor(sum, 2);
    if (q4 == 0) rinv[row] = 1.0f / sum;
  }
  __syncthreads();

  // ---- GEMM2: O = P @ V ----
  int jr = w & 1, dn = w >> 1;
  v8f o = {};
  #pragma unroll
  for (int kc = 0; kc < 8; ++kc) {
    v16bf ap = load_a16(&P[jr * 16][kc * 32], NPOS, lane);
    v16bf bv = load_b16(vT + ((size_t)(i * CDIM + h * DD + dn * 16)) * NPOS + kc * 32,
                        NPOS, lane);
    o = wmma_bf(ap, bv, o);
  }

  // ---- normalize, gate, store ----
  int col = h * DD + dn * 16 + n0;
  #pragma unroll
  for (int r = 0; r < 8; ++r) {
    int jl = jr * 16 + r + 8 * g;
    size_t m = (size_t)(i * NPOS + jt * 32 + jl);
    float gv = bf2f(gb[m * CDIM + col]);
    go[m * CDIM + col] = f2bf(o[r] * rinv[jl] * gv);
  }
}

// ---------------------------------------------------------------------------
// Kernel 5: out = go @ Wo + bo  (fp32 output)
// ---------------------------------------------------------------------------
__global__ __launch_bounds__(256) void out_kernel(
    const bf16* __restrict__ go, const bf16* __restrict__ WoT,
    const float* __restrict__ bo, float* __restrict__ out) {
  int tid = threadIdx.x, wid = tid >> 5, lane = tid & 31;
  int mw = wid & 3, nw = wid >> 2;
  size_t mt = (size_t)blockIdx.x * 64 + (size_t)mw * 16;
  int n0 = lane & 15, g = lane >> 4;

  v8f acc[4] = {};
  #pragma unroll
  for (int c = 0; c < 4; ++c) {
    v16bf a = load_a16(go + mt * CDIM + c * 32, CDIM, lane);
    #pragma unroll
    for (int t = 0; t < 4; ++t) {
      int nb = nw * 64 + t * 16;
      v16bf b = load_b16(WoT + (size_t)nb * CDIM + c * 32, CDIM, lane);
      acc[t] = wmma_bf(a, b, acc[t]);
    }
  }
  #pragma unroll
  for (int t = 0; t < 4; ++t) {
    int n = nw * 64 + t * 16 + n0;
    float bov = bo[n];
    #pragma unroll
    for (int r = 0; r < 8; ++r)
      out[(mt + r + 8 * g) * CDIM + n] = acc[t][r] + bov;
  }
}

// ---------------------------------------------------------------------------
extern "C" void kernel_launch(void* const* d_in, const int* in_sizes, int n_in,
                              void* d_out, int out_size, void* d_ws, size_t ws_size,
                              hipStream_t stream) {
  const float* pair = (const float*)d_in[0];
  const float* lns  = (const float*)d_in[1];
  const float* lnb  = (const float*)d_in[2];
  const float* Wq   = (const float*)d_in[3];
  const float* Wk   = (const float*)d_in[4];
  const float* Wv   = (const float*)d_in[5];
  const float* Wb   = (const float*)d_in[6];
  const float* Wg   = (const float*)d_in[7];
  const float* bg   = (const float*)d_in[8];
  const float* Wo   = (const float*)d_in[9];
  const float* bo   = (const float*)d_in[10];

  char* ws = (char*)d_ws;
  const size_t SZBF = (size_t)NNPOS * CDIM * sizeof(bf16);   // 16 MiB each
  bf16*  zb    = (bf16*)(ws + 0 * SZBF);
  bf16*  qb    = (bf16*)(ws + 1 * SZBF);
  bf16*  kb    = (bf16*)(ws + 2 * SZBF);
  bf16*  vT    = (bf16*)(ws + 3 * SZBF);   // [i][h*32+d][key]
  bf16*  gb    = (bf16*)(ws + 4 * SZBF);
  bf16*  go    = (bf16*)(ws + 5 * SZBF);
  float* biasb = (float*)(ws + 6 * SZBF);                    // [m][4] fp32, 1 MiB
  bf16*  wT    = (bf16*)(ws + 6 * SZBF + (size_t)NNPOS * HH * sizeof(float));

  wt_kernel<<<5 * CDIM, 128, 0, stream>>>(Wq, Wk, Wv, Wg, Wo, wT);
  ln_kernel<<<NNPOS / 8, 256, 0, stream>>>(pair, lns, lnb, Wb, zb, biasb);
  proj_kernel<<<dim3(NNPOS / 64, 4), 256, 0, stream>>>(zb, wT, bg, qb, kb, vT, gb);
  attn_kernel<<<NPOS * HH * (NPOS / 32), 128, 0, stream>>>(qb, kb, vT, gb, biasb, go);
  out_kernel<<<NNPOS / 64, 256, 0, stream>>>(go, wT + (size_t)4 * CDIM * CDIM, bo,
                                             (float*)d_out);
}
